// SelfAttention_60490319397308
// MI455X (gfx1250) — matrix operand order repaired and checked by reading, not compiled, not run on hardware
//
#include <hip/hip_runtime.h>
#include <hip/hip_bf16.h>
#include <stdint.h>

// Self-attention, B=4, S=2048, E=L=1024.
// cvt(x,W)->bf16 ; GEMM mode0 (QKV proj + bias, V transposed) ;
// GEMM mode1 (scaled Q.K^T fp32) ; row softmax (bf16 P in place) ;
// GEMM mode2 (P.V^T -> d_out fp32).
// GEMM: 128x128 block tile, TK=64, double-buffered LDS filled with
// global_load_async_to_lds_b128 (ASYNCcnt), bf16 WMMA f32 accumulate.
// Workspace (~112 MB):
//   [0,67MB)  scores fp32 (aliases x_bf16+W_bf16 early; P bf16 aliases rows)
//   [67MB) Qb bf16  [83MB) Kb bf16  [100MB) Vt bf16

#define GEMM_ASYNC 1   // flip to 0 to fall back to synchronous LDS staging

typedef __attribute__((ext_vector_type(8)))  float    v8f;
typedef __attribute__((ext_vector_type(16))) __bf16   v16bf;
typedef __attribute__((ext_vector_type(4)))  uint32_t v4u;

#define BATCH 4
#define SEQ   2048
#define LAT   1024
#define TK    64
#define LPITCH 72   // 64 k + 8 pad u16 -> 144B rows (16B aligned)

__device__ __forceinline__ uint16_t f2bf(float f) {
  uint32_t u = __float_as_uint(f);
  u += 0x7FFFu + ((u >> 16) & 1u);   // round-to-nearest-even
  return (uint16_t)(u >> 16);
}

union FragU { v4u q[2]; v16bf v; };

// Issue one stage of async tile copies: 128x64 u16 of A and of B into LDS.
// Each of the 256 threads moves 4 b128 chunks per matrix.
__device__ __forceinline__ void stage_tiles(
    const uint16_t* __restrict__ gArow, const uint16_t* __restrict__ gBrow,
    uint16_t* bufA, uint16_t* bufB, int k0, int lda, int ldb, int t)
{
  #pragma unroll
  for (int j = 0; j < 4; ++j) {
    const int chunk = t + 256 * j;       // 0..1023
    const int row   = chunk >> 3;        // 0..127
    const int col   = (chunk & 7) * 8;   // u16 units, 0..56
    const uint16_t* ga = gArow + (size_t)row * lda + k0 + col;
    const uint16_t* gb = gBrow + (size_t)row * ldb + k0 + col;
    uint16_t* la = bufA + row * LPITCH + col;
    uint16_t* lb = bufB + row * LPITCH + col;
#if GEMM_ASYNC
    uint32_t lau = (uint32_t)(size_t)la;     // low 32 bits == LDS byte address
    uint32_t lbu = (uint32_t)(size_t)lb;
    asm volatile("global_load_async_to_lds_b128 %0, %1, off"
                 :: "v"(lau), "v"((uint64_t)(size_t)ga) : "memory");
    asm volatile("global_load_async_to_lds_b128 %0, %1, off"
                 :: "v"(lbu), "v"((uint64_t)(size_t)gb) : "memory");
#else
    *(v4u*)la = *(const v4u*)ga;
    *(v4u*)lb = *(const v4u*)gb;
#endif
  }
}

__device__ __forceinline__ void wait_stage() {
#if GEMM_ASYNC
  asm volatile("s_wait_asynccnt 0x0" ::: "memory");
#endif
}

// ---------------------------------------------------------------------------
// C(MxN) = A(MxK) * B(NxK)^T, bf16 in, f32 accumulate (v_wmma_f32_16x16x32_bf16)
// ---------------------------------------------------------------------------
__global__ __launch_bounds__(256) void gemm_nt_bf16(
    const uint16_t* __restrict__ Abase, const uint16_t* __restrict__ Bbase,
    const float* __restrict__ bias,
    uint16_t* __restrict__ Qb, uint16_t* __restrict__ Kb, uint16_t* __restrict__ Vt,
    float* __restrict__ Fout,
    int mode, int Kdim, int lda, int ldb, float alpha)
{
  __shared__ __align__(16) uint16_t lsA[2][128 * LPITCH];   // 36 KB
  __shared__ __align__(16) uint16_t lsB[2][128 * LPITCH];   // 36 KB

  const size_t z = blockIdx.z;
  const uint16_t* A  = Abase;
  const uint16_t* Bm = Bbase;
  float* fout = Fout;
  if (mode == 1) {                       // scores: per-batch Q,K
    A    += z * (size_t)SEQ * LAT;
    Bm   += z * (size_t)SEQ * LAT;
    fout += z * (size_t)SEQ * SEQ;
  } else if (mode == 2) {                // out: per-batch P, V^T
    A    += z * (size_t)SEQ * 4096;      // P row stride 4096 u16 (aliased fp32 rows)
    Bm   += z * (size_t)LAT * SEQ;
    fout += z * (size_t)SEQ * LAT;
  }

  const int t    = threadIdx.x;
  const int lane = t & 31;
  const int wave = t >> 5;
  const int wr   = wave >> 2, wc = wave & 3;
  const int r0   = wr * 64,   c0 = wc * 32;
  const int lrow = lane & 15, g  = lane >> 4;

  const int blockM = blockIdx.x * 128;
  const int blockN = blockIdx.y * 128;
  const uint16_t* gArow = A  + (size_t)blockM * lda;
  const uint16_t* gBrow = Bm + (size_t)blockN * ldb;

  v8f acc[8];
  const v8f vzero = {0.f, 0.f, 0.f, 0.f, 0.f, 0.f, 0.f, 0.f};
  #pragma unroll
  for (int i = 0; i < 8; ++i) acc[i] = vzero;

  const int nst = Kdim / TK;
  stage_tiles(gArow, gBrow, lsA[0], lsB[0], 0, lda, ldb, t);

  for (int st = 0; st < nst; ++st) {
    wait_stage();          // my stage-st asyncs landed in LDS
    __syncthreads();       // everyone's landed; stage st-1 compute done by all
    if (st + 1 < nst)      // other buffer is now free: start next copy
      stage_tiles(gArow, gBrow, lsA[(st + 1) & 1], lsB[(st + 1) & 1],
                  (st + 1) * TK, lda, ldb, t);

    const uint16_t* bA = lsA[st & 1];
    const uint16_t* bB = lsB[st & 1];
    #pragma unroll
    for (int s = 0; s < 2; ++s) {        // two 32-wide K substeps
      const int ks = s * 32;
      FragU af[4], bfv[2];
      #pragma unroll
      for (int mi = 0; mi < 4; ++mi) {
        const uint16_t* p = bA + (r0 + mi * 16 + lrow) * LPITCH + ks;
        af[mi].q[0] = *(const v4u*)(p + g * 8);
        af[mi].q[1] = *(const v4u*)(p + 16 + g * 8);
      }
      #pragma unroll
      for (int ni = 0; ni < 2; ++ni) {
        const uint16_t* p = bB + (c0 + ni * 16 + lrow) * LPITCH + ks;
        bfv[ni].q[0] = *(const v4u*)(p + g * 8);
        bfv[ni].q[1] = *(const v4u*)(p + 16 + g * 8);
      }
      #pragma unroll
      for (int mi = 0; mi < 4; ++mi)
        #pragma unroll
        for (int ni = 0; ni < 2; ++ni)
          acc[mi * 2 + ni] = __builtin_amdgcn_wmma_f32_16x16x32_bf16(
              false, af[mi].v, false, bfv[ni].v,
              (short)0, acc[mi * 2 + ni], false, false);
    }
    // no trailing barrier: next iteration's leading barrier both publishes the
    // next stage and proves this stage's compute finished workgroup-wide.
  }

  // epilogue: C element (m,n): m = tile + g*8 + r, n = tile + lrow
  #pragma unroll
  for (int mi = 0; mi < 4; ++mi) {
    #pragma unroll
    for (int ni = 0; ni < 2; ++ni) {
      v8f c = acc[mi * 2 + ni];
      const int n     = blockN + c0 + ni * 16 + lrow;
      const int mBase = blockM + r0 + mi * 16 + g * 8;
      #pragma unroll
      for (int r = 0; r < 8; ++r) {
        const int m = mBase + r;
        float val = c[r];
        if (mode == 0) {
          val += bias[n];
          uint16_t h = f2bf(val);
          if (n < LAT) {
            Qb[(size_t)m * LAT + n] = h;
          } else if (n < 2 * LAT) {
            Kb[(size_t)m * LAT + (n - LAT)] = h;
          } else {                        // V stored transposed: Vt[b][d][s]
            int d  = n - 2 * LAT;
            int bb = m >> 11;
            int s  = m & 2047;
            Vt[((size_t)bb * LAT + d) * SEQ + s] = h;
          }
        } else if (mode == 1) {
          fout[(size_t)m * SEQ + n] = val * alpha;
        } else {
          fout[(size_t)m * LAT + n] = val;
        }
      }
    }
  }
}

// ---------------------------------------------------------------------------
// fp32 -> bf16, 4-wide
// ---------------------------------------------------------------------------
__global__ __launch_bounds__(256) void cvt_f32_bf16(
    const float* __restrict__ in, uint16_t* __restrict__ out, int n4)
{
  int i = blockIdx.x * blockDim.x + threadIdx.x;
  if (i < n4) {
    float4 f = ((const float4*)in)[i];
    uint64_t pk = (uint64_t)f2bf(f.x)
                | ((uint64_t)f2bf(f.y) << 16)
                | ((uint64_t)f2bf(f.z) << 32)
                | ((uint64_t)f2bf(f.w) << 48);
    ((uint64_t*)out)[i] = pk;
  }
}

// ---------------------------------------------------------------------------
// Row softmax over 2048 fp32; writes bf16 probs in place over the first half
// of each fp32 row (all reads register-resident before any write).
// ---------------------------------------------------------------------------
__global__ __launch_bounds__(256) void softmax_rows(float* __restrict__ scores)
{
  __shared__ float red[256];
  const int t = threadIdx.x;
  float* rowp = scores + (size_t)blockIdx.x * SEQ;
  uint16_t* prow = (uint16_t*)rowp;      // bf16 alias, row stride 4096 u16

  float v[8];
  #pragma unroll
  for (int j = 0; j < 8; ++j) v[j] = rowp[t + j * 256];

  float mx = v[0];
  #pragma unroll
  for (int j = 1; j < 8; ++j) mx = fmaxf(mx, v[j]);
  red[t] = mx;
  __syncthreads();
  for (int s = 128; s > 0; s >>= 1) {
    if (t < s) red[t] = fmaxf(red[t], red[t + s]);
    __syncthreads();
  }
  mx = red[0];
  __syncthreads();

  float e[8], sum = 0.f;
  #pragma unroll
  for (int j = 0; j < 8; ++j) { e[j] = __expf(v[j] - mx); sum += e[j]; }
  red[t] = sum;
  __syncthreads();
  for (int s = 128; s > 0; s >>= 1) {
    if (t < s) red[t] += red[t + s];
    __syncthreads();
  }
  const float inv = 1.0f / red[0];
  __syncthreads();

  #pragma unroll
  for (int j = 0; j < 8; ++j) prow[t + j * 256] = f2bf(e[j] * inv);
}

// ---------------------------------------------------------------------------
extern "C" void kernel_launch(void* const* d_in, const int* in_sizes, int n_in,
                              void* d_out, int out_size, void* d_ws, size_t ws_size,
                              hipStream_t stream) {
  const float* x = (const float*)d_in[0];   // (4,2048,1024)
  const float* W = (const float*)d_in[1];   // (3072,1024)
  const float* b = (const float*)d_in[2];   // (3072,)
  float* out = (float*)d_out;               // (4,2048,1024)
  char* ws = (char*)d_ws;

  uint16_t* x_bf   = (uint16_t*)ws;                          // 16 MB (dead after K1)
  uint16_t* W_bf   = (uint16_t*)(ws + (size_t)16777216);     //  6 MB (dead after K1)
  float*    scores = (float*)ws;                             // 67 MB (written in K2)
  uint16_t* P      = (uint16_t*)ws;                          // aliases score rows
  uint16_t* Qb     = (uint16_t*)(ws + (size_t)67108864);     // 16 MB
  uint16_t* Kb     = Qb + (size_t)8388608;                   // 16 MB
  uint16_t* Vt     = Kb + (size_t)8388608;                   // 16 MB (transposed V)

  // 1) fp32 -> bf16 conversions
  cvt_f32_bf16<<<dim3(8192), 256, 0, stream>>>(x, x_bf, 8388608 / 4);
  cvt_f32_bf16<<<dim3(3072), 256, 0, stream>>>(W, W_bf, 3145728 / 4);

  // 2) qkv = x @ W^T + b  (M=8192, N=3072, K=1024) -> Qb,Kb,Vt
  gemm_nt_bf16<<<dim3(64, 24, 1), 256, 0, stream>>>(
      x_bf, W_bf, b, Qb, Kb, Vt, nullptr, /*mode=*/0, 1024, 1024, 1024, 1.0f);

  // 3) scores = (1/sqrt(1024)) * Q @ K^T per batch (M=N=2048, K=1024)
  gemm_nt_bf16<<<dim3(16, 16, BATCH), 256, 0, stream>>>(
      Qb, Kb, nullptr, nullptr, nullptr, nullptr, scores,
      /*mode=*/1, 1024, 1024, 1024, 0.03125f);

  // 4) softmax over 2048-key rows, P (bf16) written in place
  softmax_rows<<<dim3(BATCH * SEQ), 256, 0, stream>>>(scores);

  // 5) out = P @ Vt^T per batch (M=2048, N=1024, K=2048)
  gemm_nt_bf16<<<dim3(16, 8, BATCH), 256, 0, stream>>>(
      P, Vt, nullptr, nullptr, nullptr, nullptr, out,
      /*mode=*/2, 2048, 4096, 2048, 1.0f);
}
